// RNNlayer_82712480186626
// MI455X (gfx1250) — matrix operand order repaired and checked by reading, not compiled, not run on hardware
//
#include <hip/hip_runtime.h>

#define B_ 128
#define T_ 1024
#define I_ 64
#define H_ 512

typedef __attribute__((ext_vector_type(16))) __bf16 v16bf;
typedef __attribute__((ext_vector_type(8)))  float  v8f;

__device__ __forceinline__ unsigned short f2bf(float f) {
  union { float f; unsigned int u; } c; c.f = f;
  unsigned int u = c.u;
  u += 0x7fffu + ((u >> 16) & 1u);          // round-to-nearest-even
  return (unsigned short)(u >> 16);
}

// ---- Pack W_hidden (f32 [H,H], row k, col n) into bf16 WMMA B-operand layout.
// half index = ((ntg*16 + kc)*32 + lane)*16 + e ; value = W[kc*32 + (e&7) + 8*(lane>>4) + 16*(e>>3)][ntg*16 + (lane&15)]
__global__ void pack_wh(const float* __restrict__ wh, unsigned short* __restrict__ whp) {
  int idx = blockIdx.x * 256 + threadIdx.x;          // 0 .. H*H-1 = 262143
  int e = idx & 15, lane = (idx >> 4) & 31, kc = (idx >> 9) & 15, ntg = idx >> 13;
  int k = kc * 32 + (e & 7) + 8 * (lane >> 4) + 16 * (e >> 3);
  int n = ntg * 16 + (lane & 15);
  whp[idx] = f2bf(wh[k * H_ + n]);
}

// ---- Pack W_in (f32 [I,H]) likewise; K=64 -> 2 K-chunks.
__global__ void pack_win(const float* __restrict__ wi, unsigned short* __restrict__ winp) {
  int idx = blockIdx.x * 256 + threadIdx.x;          // 0 .. I*H-1 = 32767
  int e = idx & 15, lane = (idx >> 4) & 31, kc = (idx >> 9) & 1, ntg = idx >> 10;
  int k = kc * 32 + (e & 7) + 8 * (lane >> 4) + 16 * (e >> 3);
  int n = ntg * 16 + (lane & 15);
  winp[idx] = f2bf(wi[k * H_ + n]);
}

// ---- out[:,0,:] = 0
__global__ void zero_t0(float* __restrict__ out) {
  int idx = blockIdx.x * 256 + threadIdx.x;          // 0 .. B*H-1
  int b = idx >> 9, h = idx & (H_ - 1);
  out[(size_t)b * T_ * H_ + h] = 0.0f;
}

// ---- Persistent scan kernel: 8 WGs (one per 16-row batch tile), 256 threads (8 waves).
// Wave w owns N-columns [w*64, w*64+64) of the 16x512 state tile.
__launch_bounds__(256, 1)
__global__ void rnn_scan(const float* __restrict__ x,
                         const float* __restrict__ sigma,
                         const unsigned short* __restrict__ whp,
                         const unsigned short* __restrict__ winp,
                         const float* __restrict__ h0,
                         float* __restrict__ out,
                         float* __restrict__ hfin) {
  // h (pre-activation) state, bf16, packed in A-matrix VGPR layout, double buffered.
  __shared__ unsigned short hb[2][16 * 32 * 16];     // 2 x 16KB
  __shared__ unsigned short xb[2][2 * 32 * 16];      // 2 x 2KB

  const int tid  = threadIdx.x;
  const int lane = tid & 31;
  const int w    = tid >> 5;
  const int b0   = blockIdx.x * 16;

  // ---- stage h0 -> hb[0] in A-packed layout (A element (l,e): m=l&15, k = (e&7)+((e>>3)<<4)+((l>>4)<<3))
  #pragma unroll
  for (int jj = 0; jj < 4; ++jj) {
    int job = tid + jj * 256;                        // 1024 jobs of 8 contiguous k
    int eh = job & 1, l = (job >> 1) & 31, kc = (job >> 6) & 15;
    int m  = l & 15;
    int ks = kc * 32 + 8 * (l >> 4) + 16 * eh;
    const float4* src = (const float4*)(h0 + (size_t)(b0 + m) * H_ + ks);
    float4 a0 = src[0], a1 = src[1];
    unsigned short* dst = &hb[0][(kc * 32 + l) * 16 + eh * 8];
    dst[0] = f2bf(a0.x); dst[1] = f2bf(a0.y); dst[2] = f2bf(a0.z); dst[3] = f2bf(a0.w);
    dst[4] = f2bf(a1.x); dst[5] = f2bf(a1.y); dst[6] = f2bf(a1.z); dst[7] = f2bf(a1.w);
  }

  v8f o[4];
  #pragma unroll
  for (int j = 0; j < 4; ++j) o[j] = v8f{0.f, 0.f, 0.f, 0.f, 0.f, 0.f, 0.f, 0.f};

  for (int t = 0; t < T_ - 1; ++t) {
    const int pb = t & 1;

    // Launder a zero *offset* (not the pointer!) so the weight loads are loop-variant
    // (no LICM hoist -> no scratch spills) while pointer provenance is preserved
    // (backend can infer global addrspace -> global_load_b128, not flat_load).
    unsigned int wofs = 0, iofs = 0;
    asm volatile("" : "+s"(wofs), "+s"(iofs));
    const v16bf* wbv = (const v16bf*)whp  + wofs;
    const v16bf* wiv = (const v16bf*)winp + iofs;

    // ---- stage x[:,t,:] tile -> xb[pb] (128 jobs)
    if (tid < 128) {
      int eh = tid & 1, l = (tid >> 1) & 31, kc = (tid >> 6) & 1;
      int m  = l & 15;
      int ks = kc * 32 + 8 * (l >> 4) + 16 * eh;
      const float4* src = (const float4*)(x + ((size_t)(b0 + m) * T_ + t) * I_ + ks);
      float4 a0 = src[0], a1 = src[1];
      unsigned short* dst = &xb[pb][(kc * 32 + l) * 16 + eh * 8];
      dst[0] = f2bf(a0.x); dst[1] = f2bf(a0.y); dst[2] = f2bf(a0.z); dst[3] = f2bf(a0.w);
      dst[4] = f2bf(a1.x); dst[5] = f2bf(a1.y); dst[6] = f2bf(a1.z); dst[7] = f2bf(a1.w);
    }
    __syncthreads();   // h_t (written last iter into hb[pb]) + x_t visible; frees hb[pb^1] for writes

    // Chunk c in [0,18): c<16 -> h_t @ W_hidden chunk c ; c>=16 -> x_t @ W_in chunk c-16.
    auto ldA = [&](int c) -> v16bf {
      if (c < 16) return *(const v16bf*)&hb[pb][(c * 32 + lane) * 16];
      return *(const v16bf*)&xb[pb][((c - 16) * 32 + lane) * 16];
    };
    auto ldB = [&](int c, int j) -> v16bf {
      int ntg = w * 4 + j;
      if (c < 16) return wbv[(ntg * 16 + c) * 32 + lane];
      return wiv[(ntg * 2 + (c - 16)) * 32 + lane];
    };

    v8f acc[4];
    #pragma unroll
    for (int j = 0; j < 4; ++j) acc[j] = v8f{0.f, 0.f, 0.f, 0.f, 0.f, 0.f, 0.f, 0.f};

    // ---- 2-stage software pipeline: prefetch A/B for chunk c+1 behind the 4 WMMAs of chunk c.
    v16bf a  = ldA(0);
    v16bf bb0 = ldB(0, 0), bb1 = ldB(0, 1), bb2 = ldB(0, 2), bb3 = ldB(0, 3);
    #pragma unroll
    for (int c = 0; c < 18; ++c) {
      v16bf an, nb0, nb1, nb2, nb3;
      if (c < 17) {
        an  = ldA(c + 1);
        nb0 = ldB(c + 1, 0); nb1 = ldB(c + 1, 1);
        nb2 = ldB(c + 1, 2); nb3 = ldB(c + 1, 3);
      }
      acc[0] = __builtin_amdgcn_wmma_f32_16x16x32_bf16(false, a, false, bb0, (short)0, acc[0], false, false);
      acc[1] = __builtin_amdgcn_wmma_f32_16x16x32_bf16(false, a, false, bb1, (short)0, acc[1], false, false);
      acc[2] = __builtin_amdgcn_wmma_f32_16x16x32_bf16(false, a, false, bb2, (short)0, acc[2], false, false);
      acc[3] = __builtin_amdgcn_wmma_f32_16x16x32_bf16(false, a, false, bb3, (short)0, acc[3], false, false);
      if (c < 17) { a = an; bb0 = nb0; bb1 = nb1; bb2 = nb2; bb3 = nb3; }
    }

    // ---- epilogue: h_new = 0.9*o + 0.1*(acc + sigma); o = tanh(h_new)
    // C/D layout: vgpr v, lane -> (m = v + 8*(lane>>4), n = ntg*16 + (lane&15))
    const int pn = pb ^ 1;
    #pragma unroll
    for (int j = 0; j < 4; ++j) {
      int ntg = w * 4 + j;
      int n   = ntg * 16 + (lane & 15);
      int hi  = lane >> 4;
      int kc2   = n >> 5, kl = n & 31;
      int ebase = (kl & 7) + 8 * (kl >> 4);
      int lbase = 8 * hi + 16 * ((kl >> 3) & 1);
      #pragma unroll
      for (int v = 0; v < 8; ++v) {
        int brow = b0 + v + 8 * hi;
        size_t gidx = ((size_t)brow * T_ + t) * H_ + n;
        float sig = sigma[gidx];
        float hn  = 0.9f * o[j][v] + 0.1f * (acc[j][v] + sig);
        float on  = tanhf(hn);
        out[gidx + H_] = on;                         // out[:, t+1, :]
        o[j][v] = on;
        hb[pn][(kc2 * 32 + (lbase + v)) * 16 + ebase] = f2bf(hn);
        if (t == T_ - 2) hfin[(size_t)brow * H_ + n] = hn;
      }
    }
  }
}

extern "C" void kernel_launch(void* const* d_in, const int* in_sizes, int n_in,
                              void* d_out, int out_size, void* d_ws, size_t ws_size,
                              hipStream_t stream) {
  const float* x  = (const float*)d_in[0];
  const float* sg = (const float*)d_in[1];
  const float* wi = (const float*)d_in[2];
  const float* wh = (const float*)d_in[3];
  const float* h0 = (const float*)d_in[4];
  float* out  = (float*)d_out;
  float* hfin = out + (size_t)B_ * T_ * H_;

  unsigned short* whp  = (unsigned short*)d_ws;      // 512 KB packed bf16 W_hidden
  unsigned short* winp = whp + (size_t)H_ * H_;      //  64 KB packed bf16 W_in

  pack_wh <<<(H_ * H_) / 256, 256, 0, stream>>>(wh, whp);
  pack_win<<<(I_ * H_) / 256, 256, 0, stream>>>(wi, winp);
  zero_t0 <<<(B_ * H_) / 256, 256, 0, stream>>>(out);
  rnn_scan<<<8, 256, 0, stream>>>(x, sg, whp, winp, h0, out, hfin);
}
